// SpikingWorker_8830452760642
// MI455X (gfx1250) — matrix operand order repaired and checked by reading, not compiled, not run on hardware
//
#include <hip/hip_runtime.h>

// SNN fused: I = x @ W^T + b (bf16 WMMA, f32 accum) fused with leaky
// integrate-and-fire scan. W (and x, if workspace allows) pre-converted to
// bf16; staging via async global->LDS when the toolchain exposes it.
// x: [B=32,S=512,H=2048] f32 ; W: [H,H] f32 ; b: [H] f32 ; out: [B,S,H] f32

#define H_DIM 2048
#define S_DIM 512
#define B_DIM 32
#define HTILE 256   // h-columns per workgroup (8 waves x 32)
#define KTILE 128   // K chunk staged in LDS
#define STILE 32    // timesteps per GEMM tile (2 WMMA M-tiles)

#if defined(__HIP_DEVICE_COMPILE__) && \
    __has_builtin(__builtin_amdgcn_global_load_async_to_lds_b128) && \
    __has_builtin(__builtin_amdgcn_s_wait_asynccnt)
#define HAVE_ASYNC 1
#else
#define HAVE_ASYNC 0
#endif

typedef __attribute__((ext_vector_type(16))) __bf16 v16bf;
typedef __attribute__((ext_vector_type(8)))  float  v8f;

union FragU {
    uint4 u[2];
    v16bf v;
};

#if HAVE_ASYNC
typedef __attribute__((__vector_size__(4 * sizeof(int)))) int i32x4;
__device__ __forceinline__ void async_cp16(const void* g, void* l) {
    __builtin_amdgcn_global_load_async_to_lds_b128(
        (i32x4*)g,
        (__attribute__((address_space(3))) i32x4*)l,
        0, 0);
}
#endif

// pack two f32 into packed bf16 dword (round-half-up) with one v_perm_b32
__device__ __forceinline__ unsigned int pack2_bf16(float a, float b) {
    unsigned int ua = __float_as_uint(a) + 0x8000u;
    unsigned int ub = __float_as_uint(b) + 0x8000u;
    return __builtin_amdgcn_perm(ub, ua, 0x07060302u);
}

// round-to-nearest-even pack (prep kernel)
__device__ __forceinline__ unsigned int pack2_bf16_rne(float a, float b) {
    unsigned int ua = __float_as_uint(a);
    unsigned int ub = __float_as_uint(b);
    ua += 0x7FFFu + ((ua >> 16) & 1u);
    ub += 0x7FFFu + ((ub >> 16) & 1u);
    return __builtin_amdgcn_perm(ub, ua, 0x07060302u);
}

// ---- prep: f32 -> bf16 bulk conversion (W and optionally x) ----
__global__ __launch_bounds__(256) void conv_bf16_kernel(const float4* __restrict__ src,
                                                        uint2* __restrict__ dst) {
    int idx = blockIdx.x * 256 + threadIdx.x;
    float4 v = src[idx];
    uint2 r;
    r.x = pack2_bf16_rne(v.x, v.y);
    r.y = pack2_bf16_rne(v.z, v.w);
    dst[idx] = r;
}

struct __align__(16) SMem {
    unsigned short sA[STILE][KTILE + 8];          // x chunk, bf16 (row = 272B)
    union {
        unsigned short sB[HTILE][KTILE + 8];      // W chunk, bf16 (live in K loop)
        float          sI[STILE][HTILE + 4];      // GEMM tile for scan (live after)
    };
};

// MODE: 0 = convert both inline, 1 = W pre-converted, 2 = W and x pre-converted
template <int MODE>
__global__ __launch_bounds__(256) void snn_fused_kernel(
    const float* __restrict__ x, const float* __restrict__ Wf,
    const unsigned short* __restrict__ Wb, const unsigned short* __restrict__ Xb,
    const float* __restrict__ bias, float* __restrict__ out) {

    __shared__ SMem sm;

    const int tid  = threadIdx.x;
    const int wave = tid >> 5;
    const int lane = tid & 31;
    const int nl   = lane & 15;
    const int hi   = lane >> 4;
    const int n0   = wave * 32;
    const int h0   = blockIdx.x * HTILE;
    const int b    = blockIdx.y;

    const float*          xb   = x  + (size_t)b * S_DIM * H_DIM;
    const unsigned short* xb16 = Xb + (size_t)b * S_DIM * H_DIM;
    float*                outb = out + (size_t)b * S_DIM * H_DIM;

    float mem = 0.0f, spk = 0.0f;
    const float bv = bias[h0 + tid];

    const int c4 = tid & 31;        // float4 col (f32 staging)
    const int r8 = tid >> 5;        // 8 row-groups (f32 staging)
    const int rB = tid >> 4;        // 16 row-groups (bf16 W staging)
    const int c8 = (tid & 15) * 8;  // bf16x8 col (bf16 W staging)
    const int rX = tid >> 3;        // 32 row-groups (bf16 x staging)
    const int cX = tid & 7;         // uint4 col (bf16 x staging)

    for (int sc = 0; sc < S_DIM / STILE; ++sc) {
        const int s0 = sc * STILE;

        v8f acc00 = {0.f,0.f,0.f,0.f,0.f,0.f,0.f,0.f};
        v8f acc01 = {0.f,0.f,0.f,0.f,0.f,0.f,0.f,0.f};
        v8f acc10 = {0.f,0.f,0.f,0.f,0.f,0.f,0.f,0.f};
        v8f acc11 = {0.f,0.f,0.f,0.f,0.f,0.f,0.f,0.f};

        for (int kc = 0; kc < H_DIM / KTILE; ++kc) {
            const int k0 = kc * KTILE;

            // ---------------- stage x tile (32 x 128 bf16) ----------------
            if constexpr (MODE == 2) {
#if HAVE_ASYNC
                #pragma unroll
                for (int p = 0; p < 2; ++p) {
                    const int c = cX + p * 8;
                    async_cp16(xb16 + (size_t)(s0 + rX) * H_DIM + k0 + c * 8,
                               &sm.sA[rX][c * 8]);
                }
#else
                #pragma unroll
                for (int p = 0; p < 2; ++p) {
                    const int c = cX + p * 8;
                    uint4 v = *(const uint4*)(xb16 + (size_t)(s0 + rX) * H_DIM + k0 + c * 8);
                    *(uint4*)&sm.sA[rX][c * 8] = v;
                }
#endif
            } else {
                #pragma unroll
                for (int p = 0; p < 4; ++p) {
                    const int r = r8 + p * 8;
                    float4 v = *(const float4*)(xb + (size_t)(s0 + r) * H_DIM + k0 + c4 * 4);
                    uint2 pk;
                    pk.x = pack2_bf16(v.x, v.y);
                    pk.y = pack2_bf16(v.z, v.w);
                    *(uint2*)&sm.sA[r][c4 * 4] = pk;
                }
            }
            // ---------------- stage W tile (256 x 128 bf16) ----------------
            if constexpr (MODE >= 1) {
#if HAVE_ASYNC
                #pragma unroll
                for (int p = 0; p < 16; ++p) {
                    const int r = rB + p * 16;
                    async_cp16(Wb + (size_t)(h0 + r) * H_DIM + k0 + c8, &sm.sB[r][c8]);
                }
#else
                #pragma unroll
                for (int p = 0; p < 16; ++p) {
                    const int r = rB + p * 16;
                    uint4 v = *(const uint4*)(Wb + (size_t)(h0 + r) * H_DIM + k0 + c8);
                    *(uint4*)&sm.sB[r][c8] = v;
                }
#endif
            } else {
                #pragma unroll
                for (int p = 0; p < 32; ++p) {
                    const int r = r8 + p * 8;
                    float4 v = *(const float4*)(Wf + (size_t)(h0 + r) * H_DIM + k0 + c4 * 4);
                    uint2 pk;
                    pk.x = pack2_bf16(v.x, v.y);
                    pk.y = pack2_bf16(v.z, v.w);
                    *(uint2*)&sm.sB[r][c4 * 4] = pk;
                }
            }
#if HAVE_ASYNC
            if constexpr (MODE >= 1) { __builtin_amdgcn_s_wait_asynccnt(0); }
#endif
            __syncthreads();

            // ---- 16x WMMA over the 128-wide K chunk: 2x2 register tile ----
            #pragma unroll
            for (int kk = 0; kk < KTILE / 32; ++kk) {
                FragU a0, a1, b0, b1;
                const unsigned short* pa0 = &sm.sA[nl][kk * 32 + hi * 8];
                a0.u[0] = *(const uint4*)(pa0);
                a0.u[1] = *(const uint4*)(pa0 + 16);
                const unsigned short* pa1 = &sm.sA[16 + nl][kk * 32 + hi * 8];
                a1.u[0] = *(const uint4*)(pa1);
                a1.u[1] = *(const uint4*)(pa1 + 16);
                const unsigned short* pb0 = &sm.sB[n0 + nl][kk * 32 + hi * 8];
                b0.u[0] = *(const uint4*)(pb0);
                b0.u[1] = *(const uint4*)(pb0 + 16);
                const unsigned short* pb1 = &sm.sB[n0 + 16 + nl][kk * 32 + hi * 8];
                b1.u[0] = *(const uint4*)(pb1);
                b1.u[1] = *(const uint4*)(pb1 + 16);

                acc00 = __builtin_amdgcn_wmma_f32_16x16x32_bf16(
                    false, a0.v, false, b0.v, (short)0, acc00, false, false);
                acc01 = __builtin_amdgcn_wmma_f32_16x16x32_bf16(
                    false, a0.v, false, b1.v, (short)0, acc01, false, false);
                acc10 = __builtin_amdgcn_wmma_f32_16x16x32_bf16(
                    false, a1.v, false, b0.v, (short)0, acc10, false, false);
                acc11 = __builtin_amdgcn_wmma_f32_16x16x32_bf16(
                    false, a1.v, false, b1.v, (short)0, acc11, false, false);
            }
            __syncthreads();
        }

        // ---- spill D tiles to sI (aliases sB; safe after last barrier) ----
        #pragma unroll
        for (int j = 0; j < 8; ++j) {
            sm.sI[j + hi * 8][n0 + nl]           = acc00[j];
            sm.sI[j + hi * 8][n0 + 16 + nl]      = acc01[j];
            sm.sI[16 + j + hi * 8][n0 + nl]      = acc10[j];
            sm.sI[16 + j + hi * 8][n0 + 16 + nl] = acc11[j];
        }
        __syncthreads();

        // ---- sequential LIF scan over 32 timesteps (all 256 threads) ----
        {
            float* orow = outb + (size_t)s0 * H_DIM + h0 + tid;
            #pragma unroll
            for (int t = 0; t < STILE; ++t) {
                float I = sm.sI[t][tid] + bv;
                mem = 0.9f * mem + I - spk;              // beta*mem + i_t - spk*thr
                float sp = (mem - 1.0f) > 0.0f ? 1.0f : 0.0f;
                orow[(size_t)t * H_DIM] = sp;
                spk = sp;
            }
        }
        __syncthreads();   // protect sI before next sc overwrites (sB alias)
    }
}

extern "C" void kernel_launch(void* const* d_in, const int* in_sizes, int n_in,
                              void* d_out, int out_size, void* d_ws, size_t ws_size,
                              hipStream_t stream) {
    (void)in_sizes; (void)n_in; (void)out_size;
    const float* x    = (const float*)d_in[0];
    const float* W    = (const float*)d_in[1];
    const float* bias = (const float*)d_in[2];
    float* out        = (float*)d_out;

    dim3 grid(H_DIM / HTILE, B_DIM);   // 8 h-tiles x 32 batches = 256 WGs
    dim3 block(256);                   // 8 waves (wave32)

    const size_t wbytes = (size_t)H_DIM * H_DIM * sizeof(unsigned short);
    const size_t xbytes = (size_t)B_DIM * S_DIM * H_DIM * sizeof(unsigned short);

    if (ws_size >= wbytes + xbytes) {
        unsigned short* Wb = (unsigned short*)d_ws;
        unsigned short* Xb = (unsigned short*)((char*)d_ws + wbytes);
        conv_bf16_kernel<<<(H_DIM * H_DIM / 4) / 256, 256, 0, stream>>>(
            (const float4*)W, (uint2*)Wb);
        conv_bf16_kernel<<<((size_t)B_DIM * S_DIM * H_DIM / 4) / 256, 256, 0, stream>>>(
            (const float4*)x, (uint2*)Xb);
        snn_fused_kernel<2><<<grid, block, 0, stream>>>(x, W, Wb, Xb, bias, out);
    } else if (ws_size >= wbytes) {
        unsigned short* Wb = (unsigned short*)d_ws;
        conv_bf16_kernel<<<(H_DIM * H_DIM / 4) / 256, 256, 0, stream>>>(
            (const float4*)W, (uint2*)Wb);
        snn_fused_kernel<1><<<grid, block, 0, stream>>>(x, W, Wb, nullptr, bias, out);
    } else {
        snn_fused_kernel<0><<<grid, block, 0, stream>>>(x, W, nullptr, nullptr, bias, out);
    }
}
